// ProsodyPredictorV15_84018150244413
// MI455X (gfx1250) — matrix operand order repaired and checked by reading, not compiled, not run on hardware
//
#include <hip/hip_runtime.h>
#include <hip/hip_bf16.h>
#include <math.h>
#include <stdint.h>

// ---------------- problem constants ----------------
#define B_    8
#define MEL_  80
#define T_    2048
#define D_    512
#define N_    16
#define NL_   6
#define LN_EPS 1e-5f
#define LC_   128              // scan chunk length
#define NC_   16               // T_/LC_ chunks
#define ASTR  100              // LDS stride (floats) for A tile: bank-conflict-free

typedef __attribute__((ext_vector_type(16))) __bf16 v16bf;
typedef __attribute__((ext_vector_type(8)))  float  v8f;

// ---- bf16 helpers (RNE) for split-precision WMMA ----
static __device__ __forceinline__ __bf16 f2bf(float f) {
  unsigned u = __float_as_uint(f);
  unsigned r = u + 0x7FFFu + ((u >> 16) & 1u);
  unsigned short s = (unsigned short)(r >> 16);
  return __builtin_bit_cast(__bf16, s);
}
static __device__ __forceinline__ float bf2f(__bf16 b) {
  unsigned short s = __builtin_bit_cast(unsigned short, b);
  return __uint_as_float(((unsigned)s) << 16);
}

static __device__ __forceinline__ float wave_sum(float v) {
  #pragma unroll
  for (int off = 16; off > 0; off >>= 1) v += __shfl_xor(v, off, 32);
  return v;
}

static __device__ __forceinline__ float gelu_tanh(float x) {
  const float k0 = 0.7978845608028654f;  // sqrt(2/pi)
  float t = tanhf(k0 * (x + 0.044715f * x * x * x));
  return 0.5f * x * (1.0f + t);
}

// ---- gfx1250 async global->LDS staging (portable inline-asm form) ----
static __device__ __forceinline__ void async_ld_b32(unsigned lds_off, const void* gaddr) {
  asm volatile("global_load_async_to_lds_b32 %0, %1, off"
               :: "v"(lds_off), "v"(gaddr) : "memory");
}
static __device__ __forceinline__ void async_ld_b128(unsigned lds_off, const void* gaddr) {
  asm volatile("global_load_async_to_lds_b128 %0, %1, off"
               :: "v"(lds_off), "v"(gaddr) : "memory");
}
static __device__ __forceinline__ void wait_async0() {
  asm volatile("s_wait_asynccnt 0x0" ::: "memory");
}
static __device__ __forceinline__ unsigned lds_off_of(const void* p) {
  // addrspace(3) flat addresses carry the LDS byte offset in the low 32 bits
  return (unsigned)(uintptr_t)p;
}

// =====================================================================
// Kernel 1: input projection, bf16x3 split-precision WMMA GEMM
//   h[b,t,:] = mel[b,:,t] @ in_w + in_b + freq_pe[t,:]
//   block = 128 thr (4 waves) -> one 16-row M-tile x four 16-col N-tiles
//   A (transposed from mel) and B staged in LDS via async global->LDS,
//   K = 80 padded to 96 (3 WMMA k-steps of 32), 3 WMMAs/k-step (bf16x3)
// =====================================================================
__global__ void __launch_bounds__(128) in_proj_wmma(
    const float* __restrict__ mel, const float* __restrict__ freq_pe,
    const float* __restrict__ in_w, const float* __restrict__ in_b,
    float* __restrict__ h) {
  __shared__ float sA[16 * ASTR];   // [m][k], k-padded, stride 100 floats
  __shared__ float sB[96 * 64];     // [k][n], 96 x 64

  const int tid   = threadIdx.x;
  const int lane  = tid & 31;
  const int wv    = tid >> 5;                 // wave -> local n-tile (0..3)
  const int mtile = blockIdx.x >> 3;          // 1024 m-tiles
  const int ng    = blockIdx.x & 7;           // 8 groups of 4 n-tiles
  const int b     = (mtile * 16) / T_;        // whole tile in one batch (T%16==0)
  const int t0    = (mtile * 16) % T_;
  const int ncol0 = ng * 64;

  // zero-fill the K-padding rows (k in [80,96)) before async staging
  for (int i = tid; i < 16 * 16; i += 128)
    sA[(i & 15) * ASTR + 80 + (i >> 4)] = 0.0f;
  for (int i = tid; i < 16 * 64; i += 128)
    sB[80 * 64 + i] = 0.0f;

  // stage A: transposing gather mel[b,k,t0+j] -> sA[j][k]  (per-lane LDS dst)
  for (int i = tid; i < MEL_ * 16; i += 128) {
    int k = i >> 4, j = i & 15;
    async_ld_b32(lds_off_of(&sA[j * ASTR + k]),
                 mel + ((size_t)(b * MEL_ + k)) * T_ + t0 + j);
  }
  // stage B: contiguous 16B rows in_w[k, ncol0+4c..] -> sB[k][4c..]
  for (int i = tid; i < MEL_ * 16; i += 128) {
    int k = i >> 4, c4 = i & 15;
    async_ld_b128(lds_off_of(&sB[k * 64 + c4 * 4]),
                  in_w + (size_t)k * D_ + ncol0 + c4 * 4);
  }
  wait_async0();
  __syncthreads();

  const int khalf = lane >> 4;
  const int ml    = lane & 15;

  v8f acc = {};
  #pragma unroll
  for (int kk = 0; kk < 96; kk += 32) {
    // A fragment (ISA 7.12.2, 16-bit A 16x32):
    //   e<8 : k = kk + e + 8*khalf ; e>=8 : k = kk + 16 + (e-8) + 8*khalf
    float av[16];
    *(float4*)&av[0]  = *(const float4*)&sA[ml * ASTR + kk + 8 * khalf];
    *(float4*)&av[4]  = *(const float4*)&sA[ml * ASTR + kk + 8 * khalf + 4];
    *(float4*)&av[8]  = *(const float4*)&sA[ml * ASTR + kk + 16 + 8 * khalf];
    *(float4*)&av[12] = *(const float4*)&sA[ml * ASTR + kk + 16 + 8 * khalf + 4];
    // B fragment: same k order as A (e<8: k=e+8*khalf; e>=8: k=16+(e-8)+8*khalf); col = wv*16+ml
    float bv[16];
    #pragma unroll
    for (int e = 0; e < 16; ++e)
      bv[e] = sB[(kk + khalf * 8 + (e & 7) + ((e >> 3) << 4)) * 64 + wv * 16 + ml];

    v16bf ahi, alo, bhi, blo;
    #pragma unroll
    for (int e = 0; e < 16; ++e) {
      __bf16 ah = f2bf(av[e]);
      ahi[e] = ah;
      alo[e] = f2bf(av[e] - bf2f(ah));
      __bf16 bh = f2bf(bv[e]);
      bhi[e] = bh;
      blo[e] = f2bf(bv[e] - bf2f(bh));
    }
    // bf16x3: hi*hi + hi*lo + lo*hi, fp32 accumulate (~fp32 product precision)
    acc = __builtin_amdgcn_wmma_f32_16x16x32_bf16(false, ahi, false, bhi, (short)0, acc, false, false);
    acc = __builtin_amdgcn_wmma_f32_16x16x32_bf16(false, ahi, false, blo, (short)0, acc, false, false);
    acc = __builtin_amdgcn_wmma_f32_16x16x32_bf16(false, alo, false, bhi, (short)0, acc, false, false);
  }

  // D layout: VGPR r -> rows r (lanes 0-15) / r+8 (lanes 16-31), col = lane%16
  #pragma unroll
  for (int r = 0; r < 8; ++r) {
    int mrow = r + 8 * khalf;
    int g    = mtile * 16 + mrow;
    int bb   = g / T_;
    int tt   = g % T_;
    int col  = ncol0 + wv * 16 + ml;
    h[(size_t)(bb * T_ + tt) * D_ + col] = acc[r] + in_b[col] + freq_pe[(size_t)tt * D_ + col];
  }
}

// =====================================================================
// Kernel 2: precompute per-layer SSM constants
//   lambda = exp(dt*A); W = 2*C*(lambda-1)/A; lambda^LC via 7 squarings
// =====================================================================
__global__ void __launch_bounds__(256) precompute_ssm(
    const float* __restrict__ log_dt, const float* __restrict__ A_re,
    const float* __restrict__ A_im, const float* __restrict__ C_re,
    const float* __restrict__ C_im,
    float* __restrict__ lam_re, float* __restrict__ lam_im,
    float* __restrict__ w_re, float* __restrict__ w_im,
    float* __restrict__ lamP_re, float* __restrict__ lamP_im) {
  int idx = blockIdx.x * blockDim.x + threadIdx.x;     // NL*D*N
  if (idx >= NL_ * D_ * N_) return;
  int ld = idx / N_;                                   // l*D + d
  float dt = expf(log_dt[ld]);
  float ar = A_re[idx], ai = A_im[idx];
  float xr = dt * ar, xi = dt * ai;
  float er = expf(xr);
  float lr = er * cosf(xi), li = er * sinf(xi);
  float nr = lr - 1.0f, ni = li;                       // Bbar = (lam-1)/A
  float inv = 1.0f / (ar * ar + ai * ai);
  float br = (nr * ar + ni * ai) * inv;
  float bi = (ni * ar - nr * ai) * inv;
  float cr = C_re[idx], ci = C_im[idx];
  float wr = 2.0f * (cr * br - ci * bi);
  float wi = 2.0f * (cr * bi + ci * br);
  float pr = lr, pi = li;                              // lam^(2^7) = lam^128
  #pragma unroll
  for (int s = 0; s < 7; ++s) {
    float qr = pr * pr - pi * pi;
    float qi = 2.0f * pr * pi;
    pr = qr; pi = qi;
  }
  lam_re[idx] = lr; lam_im[idx] = li;
  w_re[idx] = wr;  w_im[idx] = wi;
  lamP_re[idx] = pr; lamP_im[idx] = pi;
}

// =====================================================================
// Kernel 3: phase 1 of chunk-parallel scan — local chunk-final states
// =====================================================================
__global__ void __launch_bounds__(256) s4d_pass1(
    const float* __restrict__ h, const float* __restrict__ lam_re,
    const float* __restrict__ lam_im, float* __restrict__ Sloc, int layer) {
  int tid = blockIdx.x * blockDim.x + threadIdx.x;     // (b*NC + c)*D + d
  int d = tid % D_;
  int bc = tid / D_;
  int c = bc % NC_;
  int b = bc / NC_;
  const float* lre = lam_re + (layer * D_ + d) * N_;
  const float* lim = lam_im + (layer * D_ + d) * N_;
  float lr[N_], li[N_], sr[N_], si[N_];
  #pragma unroll
  for (int n = 0; n < N_; ++n) { lr[n] = lre[n]; li[n] = lim[n]; sr[n] = 0.f; si[n] = 0.f; }
  const float* up = h + ((size_t)(b * T_ + c * LC_)) * D_ + d;
  for (int t = 0; t < LC_; ++t) {
    float u = up[(size_t)t * D_];
    #pragma unroll
    for (int n = 0; n < N_; ++n) {
      float tr = sr[n];
      sr[n] = lr[n] * tr - li[n] * si[n] + u;
      si[n] = li[n] * tr + lr[n] * si[n];
    }
  }
  float* o = Sloc + (size_t)tid * (2 * N_);
  #pragma unroll
  for (int n = 0; n < N_; ++n) { o[2 * n] = sr[n]; o[2 * n + 1] = si[n]; }
}

// =====================================================================
// Kernel 4: phase 2 — sequential combine over 16 chunks per (b,d,n)
// =====================================================================
__global__ void __launch_bounds__(256) s4d_pass2(
    const float* __restrict__ Sloc, const float* __restrict__ lamP_re,
    const float* __restrict__ lamP_im, float* __restrict__ Sin, int layer) {
  int tid = blockIdx.x * blockDim.x + threadIdx.x;     // (b*D + d)*N + n
  int n = tid % N_;
  int bd = tid / N_;
  int d = bd % D_;
  int b = bd / D_;
  float pr = lamP_re[(layer * D_ + d) * N_ + n];
  float pi = lamP_im[(layer * D_ + d) * N_ + n];
  float sr = 0.f, si = 0.f;
  for (int c = 0; c < NC_; ++c) {
    size_t base = ((size_t)((b * NC_ + c) * D_ + d)) * (2 * N_);
    Sin[base + 2 * n]     = sr;
    Sin[base + 2 * n + 1] = si;
    float xr = Sloc[base + 2 * n], xi = Sloc[base + 2 * n + 1];
    float nsr = pr * sr - pi * si + xr;
    float nsi = pi * sr + pr * si + xi;
    sr = nsr; si = nsi;
  }
}

// =====================================================================
// Kernel 5: phase 3 — replay scan with correct init, emit y (in place)
// =====================================================================
__global__ void __launch_bounds__(256) s4d_pass3(
    float* __restrict__ h, const float* __restrict__ lam_re,
    const float* __restrict__ lam_im, const float* __restrict__ w_re,
    const float* __restrict__ w_im, const float* __restrict__ D_skip,
    const float* __restrict__ Sin, int layer) {
  int tid = blockIdx.x * blockDim.x + threadIdx.x;     // (b*NC + c)*D + d
  int d = tid % D_;
  int bc = tid / D_;
  int c = bc % NC_;
  int b = bc / NC_;
  const float* lre = lam_re + (layer * D_ + d) * N_;
  const float* lim = lam_im + (layer * D_ + d) * N_;
  const float* wre = w_re + (layer * D_ + d) * N_;
  const float* wim = w_im + (layer * D_ + d) * N_;
  float dskip = D_skip[layer * D_ + d];
  float lr[N_], li[N_], wr[N_], wi[N_], sr[N_], si[N_];
  const float* sp = Sin + (size_t)tid * (2 * N_);
  #pragma unroll
  for (int n = 0; n < N_; ++n) {
    lr[n] = lre[n]; li[n] = lim[n]; wr[n] = wre[n]; wi[n] = wim[n];
    sr[n] = sp[2 * n]; si[n] = sp[2 * n + 1];
  }
  float* up = h + ((size_t)(b * T_ + c * LC_)) * D_ + d;
  for (int t = 0; t < LC_; ++t) {
    float u = up[(size_t)t * D_];
    float y = 0.f;
    #pragma unroll
    for (int n = 0; n < N_; ++n) {
      float tr = sr[n];
      sr[n] = lr[n] * tr - li[n] * si[n] + u;
      si[n] = li[n] * tr + lr[n] * si[n];
      y += wr[n] * sr[n] - wi[n] * si[n];
    }
    y += dskip * u;
    up[(size_t)t * D_] = gelu_tanh(y);
  }
}

// =====================================================================
// Kernel 6: per-frame heads (f0, energy, pitch_var) — one wave per frame
// =====================================================================
__global__ void __launch_bounds__(256) frame_heads(
    const float* __restrict__ h, const float* __restrict__ hln_g,
    const float* __restrict__ hln_b, const float* __restrict__ h_w,
    const float* __restrict__ h_b, float* __restrict__ out) {
  int lane  = threadIdx.x & 31;
  int frame = blockIdx.x * 8 + (threadIdx.x >> 5);     // b*T + t
  float v[16];
  float s = 0.f;
  const float* hp = h + (size_t)frame * D_ + lane * 16;
  #pragma unroll
  for (int j = 0; j < 16; ++j) { v[j] = hp[j]; s += v[j]; }
  float mu = wave_sum(s) * (1.0f / D_);
  float q = 0.f;
  #pragma unroll
  for (int j = 0; j < 16; ++j) { float dd = v[j] - mu; q += dd * dd; }
  float var = wave_sum(q) * (1.0f / D_);
  float inv = rsqrtf(var + LN_EPS);
  #pragma unroll
  for (int head = 0; head < 3; ++head) {
    float p = 0.f;
    #pragma unroll
    for (int j = 0; j < 16; ++j) {
      int d = lane * 16 + j;
      p += ((v[j] - mu) * inv * hln_g[head * D_ + d] + hln_b[head * D_ + d]) * h_w[head * D_ + d];
    }
    p = wave_sum(p);
    if (lane == 0) out[head * (B_ * T_) + frame] = p + h_b[head];
  }
}

// =====================================================================
// Kernel 7: time mean hm[b,d] = mean_t h[b,t,d]
// =====================================================================
__global__ void __launch_bounds__(256) time_mean(
    const float* __restrict__ h, float* __restrict__ hm) {
  int tid = blockIdx.x * blockDim.x + threadIdx.x;     // b*D + d
  int d = tid % D_;
  int b = tid / D_;
  float s = 0.f;
  const float* p = h + (size_t)b * T_ * D_ + d;
  for (int t = 0; t < T_; ++t) s += p[(size_t)t * D_];
  hm[tid] = s * (1.0f / T_);
}

// =====================================================================
// Kernel 8: utterance heads (speech_rate, pause_dur, mfcc) — wave per b
// =====================================================================
__global__ void __launch_bounds__(256) utt_heads(
    const float* __restrict__ hm, const float* __restrict__ hln_g,
    const float* __restrict__ hln_b, const float* __restrict__ h_w,
    const float* __restrict__ h_b, const float* __restrict__ mf_g,
    const float* __restrict__ mf_b, const float* __restrict__ mf_w,
    const float* __restrict__ mf_b2, float* __restrict__ out) {
  int lane = threadIdx.x & 31;
  int b    = threadIdx.x >> 5;                         // 8 waves = 8 batches
  float v[16];
  float s = 0.f;
  const float* hp = hm + b * D_ + lane * 16;
  #pragma unroll
  for (int j = 0; j < 16; ++j) { v[j] = hp[j]; s += v[j]; }
  float mu = wave_sum(s) * (1.0f / D_);
  float q = 0.f;
  #pragma unroll
  for (int j = 0; j < 16; ++j) { float dd = v[j] - mu; q += dd * dd; }
  float var = wave_sum(q) * (1.0f / D_);
  float inv = rsqrtf(var + LN_EPS);
  const int base = 3 * B_ * T_;
  #pragma unroll
  for (int hh = 0; hh < 2; ++hh) {                     // heads 3,4
    int head = 3 + hh;
    float p = 0.f;
    #pragma unroll
    for (int j = 0; j < 16; ++j) {
      int d = lane * 16 + j;
      p += ((v[j] - mu) * inv * hln_g[head * D_ + d] + hln_b[head * D_ + d]) * h_w[head * D_ + d];
    }
    p = wave_sum(p);
    if (lane == 0) out[base + hh * B_ + b] = p + h_b[head];
  }
  for (int k = 0; k < 13; ++k) {                       // mfcc
    float p = 0.f;
    #pragma unroll
    for (int j = 0; j < 16; ++j) {
      int d = lane * 16 + j;
      p += ((v[j] - mu) * inv * mf_g[d] + mf_b[d]) * mf_w[d * 13 + k];
    }
    p = wave_sum(p);
    if (lane == 0) out[base + 2 * B_ + b * 13 + k] = p + mf_b2[k];
  }
}

// =====================================================================
extern "C" void kernel_launch(void* const* d_in, const int* in_sizes, int n_in,
                              void* d_out, int out_size, void* d_ws, size_t ws_size,
                              hipStream_t stream) {
  const float* mel     = (const float*)d_in[0];
  const float* freq_pe = (const float*)d_in[1];
  const float* in_w    = (const float*)d_in[2];
  const float* in_b    = (const float*)d_in[3];
  const float* log_dt  = (const float*)d_in[4];
  const float* A_re    = (const float*)d_in[5];
  const float* A_im    = (const float*)d_in[6];
  const float* C_re    = (const float*)d_in[7];
  const float* C_im    = (const float*)d_in[8];
  const float* D_skip  = (const float*)d_in[9];
  const float* hln_g   = (const float*)d_in[10];
  const float* hln_b   = (const float*)d_in[11];
  const float* h_w     = (const float*)d_in[12];
  const float* h_b     = (const float*)d_in[13];
  const float* mf_g    = (const float*)d_in[14];
  const float* mf_b    = (const float*)d_in[15];
  const float* mf_w    = (const float*)d_in[16];
  const float* mf_b2   = (const float*)d_in[17];

  // workspace carve-up (floats): ~52 MB total
  float* ws      = (float*)d_ws;
  float* h       = ws;                                  // B*T*D = 8388608
  float* lam_re  = h + (size_t)B_ * T_ * D_;
  float* lam_im  = lam_re + NL_ * D_ * N_;
  float* w_re    = lam_im + NL_ * D_ * N_;
  float* w_im    = w_re + NL_ * D_ * N_;
  float* lamP_re = w_im + NL_ * D_ * N_;
  float* lamP_im = lamP_re + NL_ * D_ * N_;
  float* Sloc    = lamP_im + NL_ * D_ * N_;             // B*NC*D*2N = 2097152
  float* Sin     = Sloc + (size_t)B_ * NC_ * D_ * 2 * N_;
  float* hm      = Sin + (size_t)B_ * NC_ * D_ * 2 * N_; // B*D

  // 1) input projection GEMM (async-LDS staged, WMMA bf16x3) + bias + pos-emb
  //    1024 m-tiles x 8 n-groups, 128 threads (4 waves = 4 n-tiles) each
  in_proj_wmma<<<(B_ * T_ / 16) * (D_ / 64), 128, 0, stream>>>(
      mel, freq_pe, in_w, in_b, h);

  // 2) SSM constants for all layers
  precompute_ssm<<<(NL_ * D_ * N_ + 255) / 256, 256, 0, stream>>>(
      log_dt, A_re, A_im, C_re, C_im,
      lam_re, lam_im, w_re, w_im, lamP_re, lamP_im);

  // 3) six S4D layers as chunk-parallel linear-recurrence scans (in place)
  const int scan_blocks = (B_ * NC_ * D_) / 256;        // 256
  const int comb_blocks = (B_ * D_ * N_) / 256;         // 256
  for (int l = 0; l < NL_; ++l) {
    s4d_pass1<<<scan_blocks, 256, 0, stream>>>(h, lam_re, lam_im, Sloc, l);
    s4d_pass2<<<comb_blocks, 256, 0, stream>>>(Sloc, lamP_re, lamP_im, Sin, l);
    s4d_pass3<<<scan_blocks, 256, 0, stream>>>(h, lam_re, lam_im, w_re, w_im,
                                               D_skip, Sin, l);
  }

  // 4) heads
  frame_heads<<<(B_ * T_) / 8, 256, 0, stream>>>(h, hln_g, hln_b, h_w, h_b,
                                                 (float*)d_out);
  time_mean<<<(B_ * D_) / 256, 256, 0, stream>>>(h, hm);
  utt_heads<<<1, 256, 0, stream>>>(hm, hln_g, hln_b, h_w, h_b,
                                   mf_g, mf_b, mf_w, mf_b2, (float*)d_out);
}
